// BRC_88587995447900
// MI455X (gfx1250) — matrix operand order repaired and checked by reading, not compiled, not run on hardware
//
#include <hip/hip_runtime.h>

// BRC (bistable recurrent cell), 2 layers, S=512 B=128 IN=H=256.
//
// Phase 1 per layer: fused projection GEMM (M=65536, N=256, K=256) for
//   xc/xa/xh via v_wmma_f32_16x16x32_bf16 (fp32 accumulate).
//   - Each block: 4 waves own a 64-wide N slice and 16 M-tiles (256 rows).
//   - U{c,a,h} slices staged ONCE into LDS, as T[n_local][k] bf16 (k contiguous)
//     (3 x 32KB = 96KB -> 3 blocks/WGP in the 320KB WGP LDS).
//   - Inner loop: A fragment = 4x global_load_b128 + packed bf16 converts,
//     shared across the three matrices; B fragments = one 32B LDS read each;
//     K loop fully unrolled -> 24 static WMMAs per m-tile.
//   - Output packed as {bf16 xc, xa, xh, 0} in 8 bytes -> one b64 store/row.
//
// Phase 2 per layer: persistent elementwise scan, h register-resident for all
//   512 steps; ONE global_load_b64 per step (packed triple), software
//   pipelined one step ahead + global_prefetch two steps ahead. The 128MB
//   packed projection buffer fits the 192MB L2.
//
// Workspace: 512*128*256 * 8B = 128 MB in d_ws.

typedef __attribute__((ext_vector_type(16))) __bf16 v16bf;
typedef __attribute__((ext_vector_type(8)))  float  v8f;
typedef __attribute__((ext_vector_type(4)))  float  v4f;

constexpr int S_LEN = 512;
constexpr int B_SZ  = 128;
constexpr int K_IN  = 256;
constexpr int H_DIM = 256;
constexpr int L_NUM = 2;
constexpr int MT    = 16;   // m-tiles per block

__device__ __forceinline__ unsigned pack2bf(float lo, float hi) {
  const __bf16 a = (__bf16)lo, b = (__bf16)hi;
  const unsigned short ua = __builtin_bit_cast(unsigned short, a);
  const unsigned short ub = __builtin_bit_cast(unsigned short, b);
  return ((unsigned)ub << 16) | (unsigned)ua;
}
__device__ __forceinline__ float bf_lo(unsigned u) {
  return __builtin_bit_cast(float, u << 16);
}
__device__ __forceinline__ float bf_hi(unsigned u) {
  return __builtin_bit_cast(float, u & 0xffff0000u);
}

// ---------------------------------------------------------------------------
// Fused projection GEMM. grid = (M/(16*MT), H/64) = (256, 4), block = 128.
// ---------------------------------------------------------------------------
__global__ __launch_bounds__(128, 3)
void brc_proj_kernel(const float* __restrict__ X,    // (M, K) row-major
                     const float* __restrict__ Uc,   // (H, K) row-major
                     const float* __restrict__ Ua,
                     const float* __restrict__ Uh,
                     const float* __restrict__ bc,   // (H) layer slice
                     const float* __restrict__ ba,
                     uint2* __restrict__ P)          // (M, H) packed triples
{
  __shared__ __align__(32) __bf16 Tc[K_IN * 64];   // T[n_local][k]
  __shared__ __align__(32) __bf16 Ta[K_IN * 64];
  __shared__ __align__(32) __bf16 Th[K_IN * 64];

  const int tid   = threadIdx.x;
  const int lane  = tid & 31;
  const int wave  = tid >> 5;
  const int hi    = lane >> 4;     // lane half
  const int l15   = lane & 15;
  const int nBase = blockIdx.y * 64;

  // ---- stage U slices (rows nBase..+63, all K), k-contiguous, bf16 ----
  {
    const int r  = tid >> 1;           // n_local row 0..63
    const int c0 = (tid & 1) * 128;    // k half
    const float* rc = Uc + (size_t)(nBase + r) * K_IN + c0;
    const float* ra = Ua + (size_t)(nBase + r) * K_IN + c0;
    const float* rh = Uh + (size_t)(nBase + r) * K_IN + c0;
#pragma unroll 4
    for (int i = 0; i < 128; i += 4) {
      const v4f vc = *(const v4f*)(rc + i);
      const v4f va = *(const v4f*)(ra + i);
      const v4f vh = *(const v4f*)(rh + i);
#pragma unroll
      for (int j = 0; j < 4; ++j) {
        const int k = c0 + i + j;
        Tc[r * K_IN + k] = (__bf16)vc[j];
        Ta[r * K_IN + k] = (__bf16)va[j];
        Th[r * K_IN + k] = (__bf16)vh[j];
      }
    }
  }
  __syncthreads();

  const int   n0l   = wave * 16;             // n_local tile base
  const int   col   = nBase + n0l + l15;     // output column for this lane
  const float biasC = bc[col];
  const float biasA = ba[col];
  const int   mBase = blockIdx.x * (16 * MT);

#pragma unroll 1
  for (int mt = 0; mt < MT; ++mt) {
    const int m0 = mBase + mt * 16;
    const float* Xrow = X + (size_t)(m0 + l15) * K_IN;

    v8f accC = {};
    v8f accA = {};
    v8f accH = {};

#pragma unroll
    for (int kc = 0; kc < 8; ++kc) {
      const int k0 = kc * 32;
      // A fragment: one contiguous 16-float run per lane (k = 16*hi + s)
      const v4f x0 = *(const v4f*)(Xrow + k0 + hi * 16);
      const v4f x1 = *(const v4f*)(Xrow + k0 + hi * 16 + 4);
      const v4f x2 = *(const v4f*)(Xrow + k0 + hi * 16 + 8);
      const v4f x3 = *(const v4f*)(Xrow + k0 + hi * 16 + 12);
      v16bf af;
#pragma unroll
      for (int j = 0; j < 4; ++j) {
        af[j]      = (__bf16)x0[j];
        af[4 + j]  = (__bf16)x1[j];
        af[8 + j]  = (__bf16)x2[j];
        af[12 + j] = (__bf16)x3[j];
      }
      // B fragments: one aligned 32B LDS read each (lane -> N column)
      const int kb = (n0l + l15) * K_IN + k0 + 16 * hi;
      const v16bf bC = *(const v16bf*)&Tc[kb];
      const v16bf bA = *(const v16bf*)&Ta[kb];
      const v16bf bH = *(const v16bf*)&Th[kb];

      accC = __builtin_amdgcn_wmma_f32_16x16x32_bf16(false, af, false, bC,
                                                     (short)0, accC, false, false);
      accA = __builtin_amdgcn_wmma_f32_16x16x32_bf16(false, af, false, bA,
                                                     (short)0, accA, false, false);
      accH = __builtin_amdgcn_wmma_f32_16x16x32_bf16(false, af, false, bH,
                                                     (short)0, accH, false, false);
    }

    // C/D layout: VGPR r, lane l -> M = r + 8*(l>>4), N = l&15
#pragma unroll
    for (int r = 0; r < 8; ++r) {
      const size_t off = (size_t)(m0 + r + 8 * hi) * H_DIM + col;
      uint2 pk;
      pk.x = pack2bf(accC[r] + biasC, accA[r] + biasA);
      pk.y = pack2bf(accH[r], 0.0f);
      P[off] = pk;
    }
  }
}

// ---------------------------------------------------------------------------
// Persistent scan: one thread per (b, j); one b64 load per step.
// grid = B*H/256 = 128 blocks, block = 256.
// ---------------------------------------------------------------------------
__global__ __launch_bounds__(256, 4)
void brc_scan_kernel(const uint2* __restrict__ P,
                     const float* __restrict__ h0,   // (B*H) layer slice
                     const float* __restrict__ wc,   // (H) layer slice
                     const float* __restrict__ wa,
                     float* __restrict__ out,        // (S, B*H)
                     float* __restrict__ hn)         // (B*H) layer slice
{
  constexpr int STRIDE = B_SZ * H_DIM;
  const int idx = blockIdx.x * blockDim.x + threadIdx.x;
  const int j   = idx & (H_DIM - 1);

  float h         = h0[idx];
  const float wcv = wc[j];
  const float wav = wa[j];

  uint2 p = P[idx];   // t = 0

#pragma unroll 1
  for (int t = 0; t < S_LEN; ++t) {
    uint2 np = {0u, 0u};
    if (t + 1 < S_LEN) np = P[(size_t)(t + 1) * STRIDE + idx];
    if (t + 2 < S_LEN)
      __builtin_prefetch(P + (size_t)(t + 2) * STRIDE + idx, 0, 1);

    const float xcv = bf_lo(p.x);
    const float xav = bf_hi(p.x);
    const float xhv = bf_lo(p.y);

    const float zc   = xcv + wcv * h;
    const float c    = 1.f / (1.f + __expf(-zc));     // sigmoid
    const float a    = 1.f + tanhf(xav + wav * h);
    const float cand = tanhf(xhv + a * h);
    h = c * h + (1.f - c) * cand;

    out[(size_t)t * STRIDE + idx] = h;
    p = np;
  }
  hn[idx] = h;
}

// ---------------------------------------------------------------------------
extern "C" void kernel_launch(void* const* d_in, const int* in_sizes, int n_in,
                              void* d_out, int out_size, void* d_ws, size_t ws_size,
                              hipStream_t stream)
{
  (void)in_sizes; (void)n_in; (void)out_size; (void)ws_size;

  const float* x  = (const float*)d_in[0];  // (S, B, IN)
  const float* h0 = (const float*)d_in[1];  // (L, B, H)
  const float* Uc = (const float*)d_in[2];  // (L, H, IN)
  const float* wc = (const float*)d_in[3];  // (L, H)
  const float* bc = (const float*)d_in[4];  // (L, H)
  const float* Ua = (const float*)d_in[5];
  const float* wa = (const float*)d_in[6];
  const float* ba = (const float*)d_in[7];
  const float* Uh = (const float*)d_in[8];

  const size_t SBH = (size_t)S_LEN * B_SZ * H_DIM;   // 16,777,216
  const size_t BH  = (size_t)B_SZ * H_DIM;           // 32,768

  float* out = (float*)d_out;        // (S,B,H) final-layer sequence output
  float* hn  = out + SBH;            // (L,B,H) stacked final hidden states

  uint2* P = (uint2*)d_ws;           // 128MB packed projection triples

  const dim3 gemm_grid(S_LEN * B_SZ / (16 * MT), H_DIM / 64);  // (256, 4)
  const dim3 scan_grid(BH / 256);                              // 128 blocks

  for (int l = 0; l < L_NUM; ++l) {
    // Layer input: original x for layer 0; layer-0 sequence output (staged in
    // d_out) for layer 1. The GEMM fully consumes it before the scan
    // overwrites d_out (same stream => ordered).
    const float* Xin = (l == 0) ? x : out;
    const size_t uoff = (size_t)l * H_DIM * K_IN;
    const size_t voff = (size_t)l * H_DIM;

    brc_proj_kernel<<<gemm_grid, 128, 0, stream>>>(
        Xin, Uc + uoff, Ua + uoff, Uh + uoff, bc + voff, ba + voff, P);

    brc_scan_kernel<<<scan_grid, 256, 0, stream>>>(
        P, h0 + (size_t)l * BH, wc + voff, wa + voff,
        out, hn + (size_t)l * BH);
  }
}
